// FixPointLayer_28527172780283
// MI455X (gfx1250) — compile-verified, matrix operand order checked by reading, not run on hardware
//
#include <hip/hip_runtime.h>
#include <hip/hip_bf16.h>

// ---------------------------------------------------------------------------
// Fixed-point layer: z <- tanh(z @ W^T + x), 50 steps.
//   x : [8192, 1024] f32   (d_in[0])
//   W : [1024, 1024] f32   (d_in[1])
//   out: [8192, 1024] f32  (d_out)
// bf16 WMMA with f32 accumulation; z kept in bf16 ping-pong buffers in d_ws.
// Tile copies use GLOBAL_LOAD_ASYNC_TO_LDS_B128 (ASYNCcnt) when the builtin
// exists, else a global->VGPR->ds_store fallback.
// ---------------------------------------------------------------------------

typedef __attribute__((ext_vector_type(16))) __bf16 v16bf;
typedef __attribute__((ext_vector_type(8)))  __bf16 bf16x8;
typedef __attribute__((ext_vector_type(8)))  float  v8f;

#define B_ROWS 8192
#define FDIM   1024
#define BM     128
#define BN     128
#define BK     32
#define LDSK   40              // padded LDS row stride (80B, 16B aligned, conflict-free)
#define NKT    (FDIM / BK)     // 32 k-steps

#if defined(__has_builtin)
#if __has_builtin(__builtin_amdgcn_global_load_async_to_lds_b128)
#define USE_ASYNC_COPY 1
#endif
#endif

#ifdef USE_ASYNC_COPY
// Exact pointee type from the clang diagnostic:
//   "__attribute__((__vector_size__(4 * sizeof(int)))) int __device__ *"
typedef int v4i_vec __attribute__((__vector_size__(16)));
typedef __attribute__((address_space(1))) v4i_vec* gbl_v4i_ptr;   // global (AS1)
typedef __attribute__((address_space(3))) v4i_vec* lds_v4i_ptr;   // LDS (AS3)
#endif

__device__ __forceinline__ float fast_tanh(float x) {
#if __has_builtin(__builtin_amdgcn_tanhf)
    return __builtin_amdgcn_tanhf(x);
#elif __has_builtin(__builtin_amdgcn_tanh_f32)
    return __builtin_amdgcn_tanh_f32(x);
#else
    return tanhf(x);
#endif
}

#ifdef USE_ASYNC_COPY
__device__ __forceinline__ void wait_async0() {
#if __has_builtin(__builtin_amdgcn_s_wait_asynccnt)
    __builtin_amdgcn_s_wait_asynccnt(0);
#else
    asm volatile("s_wait_asynccnt 0x0" ::: "memory");
#endif
}
#endif

// Step 1: z1 = tanh(0 @ W^T + x) = tanh(x), stored as bf16.
__global__ void fixpoint_init_kernel(const float* __restrict__ x,
                                     __bf16* __restrict__ z, int n) {
    int i = blockIdx.x * blockDim.x + threadIdx.x;
    if (i < n) z[i] = (__bf16)fast_tanh(x[i]);
}

__global__ void convert_w_kernel(const float* __restrict__ W,
                                 __bf16* __restrict__ Wb, int n) {
    int i = blockIdx.x * blockDim.x + threadIdx.x;
    if (i < n) Wb[i] = (__bf16)W[i];
}

// One fixed-point step: zout = tanh(zin @ W^T + x).
// outf != nullptr on the final step -> also emit f32 result.
__global__ __launch_bounds__(256)
void fixpoint_gemm_kernel(const __bf16* __restrict__ zin,  // [8192,1024] bf16
                          const __bf16* __restrict__ Wb,   // [1024,1024] bf16 (row n = W[n,:])
                          const float*  __restrict__ x,    // [8192,1024] f32
                          __bf16* __restrict__ zout,       // [8192,1024] bf16
                          float*  __restrict__ outf)       // optional f32 out
{
    __shared__ __bf16 sA[2][BM * LDSK];   // z tile:  row m -> K slice
    __shared__ __bf16 sB[2][BN * LDSK];   // W tile:  row n -> K slice (col n of z@W^T)

    const int tid   = threadIdx.x;
    const int lane  = tid & 31;
    const int wave  = tid >> 5;          // 0..7
    const int waveM = wave & 1;          // 64-row half of the 128-row tile
    const int waveN = wave >> 1;         // 32-col slice of the 128-col tile

    const int rowBase = blockIdx.x * BM;
    const int colBase = blockIdx.y * BN;

    const int hi   = lane >> 4;          // 0 / 1
    const int lidx = lane & 15;          // row (A) / col (B) within 16
    const int koff = hi * 8;             // ISA 16-bit fragment K pattern

    v8f acc[4][2] = {};

    auto copy_tile = [&](int buf, int k0) {
        // A tile + B tile, 512 x 16B chunks each; 256 lanes, 2 chunks apiece.
        for (int c = tid; c < 512; c += 256) {
            const int row = c >> 2;      // 0..127
            const int seg = c & 3;       // 4 x (8 bf16) per row
            const __bf16* ga = zin + (size_t)(rowBase + row) * FDIM + k0 + seg * 8;
            const __bf16* gb = Wb  + (size_t)(colBase + row) * FDIM + k0 + seg * 8;
            __bf16* la = &sA[buf][row * LDSK + seg * 8];
            __bf16* lb = &sB[buf][row * LDSK + seg * 8];
#ifdef USE_ASYNC_COPY
            __builtin_amdgcn_global_load_async_to_lds_b128(
                (gbl_v4i_ptr)(void*)ga, (lds_v4i_ptr)(void*)la, 0, 0);
            __builtin_amdgcn_global_load_async_to_lds_b128(
                (gbl_v4i_ptr)(void*)gb, (lds_v4i_ptr)(void*)lb, 0, 0);
#else
            *(bf16x8*)la = *(const bf16x8*)ga;
            *(bf16x8*)lb = *(const bf16x8*)gb;
#endif
        }
        // Nudge the k+2 slice toward the caches (global_prefetch_b8).
        if (k0 + 2 * BK < FDIM && tid < 128) {
            __builtin_prefetch(zin + (size_t)(rowBase + tid) * FDIM + k0 + 2 * BK, 0, 0);
            __builtin_prefetch(Wb  + (size_t)(colBase + tid) * FDIM + k0 + 2 * BK, 0, 0);
        }
    };

    copy_tile(0, 0);

    for (int kt = 0; kt < NKT; ++kt) {
#ifdef USE_ASYNC_COPY
        wait_async0();                   // my async writes to LDS are done
#endif
        __syncthreads();                 // tile kt visible; buf (kt+1)&1 free
        if (kt + 1 < NKT) copy_tile((kt + 1) & 1, (kt + 1) * BK);

        const int buf = kt & 1;
        v16bf afrag[4], bfrag[2];
        #pragma unroll
        for (int mt = 0; mt < 4; ++mt) {
            const __bf16* p = &sA[buf][(waveM * 64 + mt * 16 + lidx) * LDSK + koff];
            ((bf16x8*)&afrag[mt])[0] = *(const bf16x8*)p;         // K koff..koff+7
            ((bf16x8*)&afrag[mt])[1] = *(const bf16x8*)(p + 16);  // K koff+16..koff+23
        }
        #pragma unroll
        for (int nt = 0; nt < 2; ++nt) {
            const __bf16* p = &sB[buf][(waveN * 32 + nt * 16 + lidx) * LDSK + koff];
            ((bf16x8*)&bfrag[nt])[0] = *(const bf16x8*)p;
            ((bf16x8*)&bfrag[nt])[1] = *(const bf16x8*)(p + 16);
        }

        #pragma unroll
        for (int mt = 0; mt < 4; ++mt)
            #pragma unroll
            for (int nt = 0; nt < 2; ++nt)
                acc[mt][nt] = __builtin_amdgcn_wmma_f32_16x16x32_bf16(
                    /*neg_a=*/false, afrag[mt],
                    /*neg_b=*/false, bfrag[nt],
                    /*c_mod=*/(short)0, acc[mt][nt],
                    /*reuse_a=*/false, /*reuse_b=*/false);
    }

    // Epilogue: D-layout lane<16 -> M=v, lane>=16 -> M=v+8; N = lane&15.
    #pragma unroll
    for (int mt = 0; mt < 4; ++mt) {
        #pragma unroll
        for (int nt = 0; nt < 2; ++nt) {
            #pragma unroll
            for (int v = 0; v < 8; ++v) {
                const int gr = rowBase + waveM * 64 + mt * 16 + hi * 8 + v;
                const int gc = colBase + waveN * 32 + nt * 16 + lidx;
                const size_t idx = (size_t)gr * FDIM + gc;
                const float val = fast_tanh(acc[mt][nt][v] + x[idx]);
                zout[idx] = (__bf16)val;
                if (outf) outf[idx] = val;
            }
        }
    }
}

extern "C" void kernel_launch(void* const* d_in, const int* in_sizes, int n_in,
                              void* d_out, int out_size, void* d_ws, size_t ws_size,
                              hipStream_t stream) {
    const float* x = (const float*)d_in[0];   // [8192,1024]
    const float* W = (const float*)d_in[1];   // [1024,1024]
    float* out = (float*)d_out;

    char* ws = (char*)d_ws;
    __bf16* Wb = (__bf16*)ws;                                               // 2 MB
    __bf16* zA = (__bf16*)(ws + (size_t)FDIM * FDIM * sizeof(__bf16));      // 16 MB
    __bf16* zB = (__bf16*)(ws + (size_t)FDIM * FDIM * sizeof(__bf16)
                              + (size_t)B_ROWS * FDIM * sizeof(__bf16));    // 16 MB

    const int nW = FDIM * FDIM;
    const int nZ = B_ROWS * FDIM;
    convert_w_kernel<<<(nW + 255) / 256, 256, 0, stream>>>(W, Wb, nW);
    fixpoint_init_kernel<<<(nZ + 255) / 256, 256, 0, stream>>>(x, zA, nZ);

    dim3 grid(B_ROWS / BM, FDIM / BN);        // 64 x 8 workgroups
    __bf16* zin = zA;
    __bf16* zout = zB;
    for (int it = 0; it < 49; ++it) {         // steps 2..50
        float* of = (it == 48) ? out : nullptr;
        fixpoint_gemm_kernel<<<grid, 256, 0, stream>>>(zin, Wb, x, zout, of);
        __bf16* t = zin; zin = zout; zout = t;
    }
}
